// CAA_Module_16922171146623
// MI455X (gfx1250) — matrix operand order repaired and checked by reading, not compiled, not run on hardware
//
#include <hip/hip_runtime.h>
#include <hip/hip_bf16.h>

#define BB 32
#define CC 1024
#define NN 256
#define DQH 64
#define EPSF 1e-5f

typedef __attribute__((ext_vector_type(16))) __bf16 v16bf;
typedef __attribute__((ext_vector_type(8)))  float  v8f;

__device__ __forceinline__ v8f wmma_bf16(v16bf a, v16bf b, v8f c) {
  // D = A(16x32) * B(32x16) + C, f32 accumulate
  return __builtin_amdgcn_wmma_f32_16x16x32_bf16(false, a, false, b, (short)0, c, false, false);
}

// ---- Fragment loaders (bf16 sources only; conversions are done once up-front)
// A fragment (16x32, MxK) from row-major MxK bf16; ld in elements.
// ISA 16-bit A layout: lane l: m=l&15; element j -> k = ((j&8)<<1) + 8*(l>>4) + (j&7)
// Per lane this is two contiguous 8-element (16B) runs -> vectorizes to b128 loads.
__device__ __forceinline__ v16bf fragA_bf16(const __bf16* __restrict__ src, int ld) {
  const int l = threadIdx.x & 31;
  const __bf16* row = src + (l & 15) * ld + 8 * (l >> 4);
  v16bf a;
#pragma unroll
  for (int j = 0; j < 16; ++j) a[j] = row[((j & 8) << 1) + (j & 7)];
  return a;
}
// B fragment (32x16, KxN) from Bt storage (N x K row-major, k contiguous).
// ISA 16-bit B layout: lane l: n=l&15; element j -> k = ((j&8)<<1) + 8*(l>>4) + (j&7)
__device__ __forceinline__ v16bf fragB_bf16(const __bf16* __restrict__ src, int ld) {
  const int l = threadIdx.x & 31;
  const __bf16* p = src + (l & 15) * ld + 8 * (l >> 4);
  v16bf b;
#pragma unroll
  for (int j = 0; j < 16; ++j) b[j] = p[((j & 8) << 1) + (j & 7)];
  return b;
}

// ---- Kernel 0: one-pass f32 -> bf16 conversion of x, Wv, Wq, Wk -------------
#define XE  ((size_t)BB * CC * NN)   // 8,388,608
#define WVE ((size_t)CC * CC)        // 1,048,576
#define WQE ((size_t)DQH * NN)       // 16,384
__global__ void cvt_kernel(const float* __restrict__ x,  const float* __restrict__ Wv,
                           const float* __restrict__ Wq, const float* __restrict__ Wk,
                           __bf16* __restrict__ Xb, __bf16* __restrict__ Wvb,
                           __bf16* __restrict__ Wqb, __bf16* __restrict__ Wkb) {
  const size_t i4 = ((size_t)blockIdx.x * blockDim.x + threadIdx.x) * 4;
  const float* src; __bf16* dst; size_t off;
  if (i4 < XE)                       { src = x;  dst = Xb;  off = i4; }
  else if (i4 < XE + WVE)            { src = Wv; dst = Wvb; off = i4 - XE; }
  else if (i4 < XE + WVE + WQE)      { src = Wq; dst = Wqb; off = i4 - XE - WVE; }
  else if (i4 < XE + WVE + 2 * WQE)  { src = Wk; dst = Wkb; off = i4 - XE - WVE - WQE; }
  else return;
  const float4 v = *(const float4*)(src + off);
  __bf16 o4[4] = {(__bf16)v.x, (__bf16)v.y, (__bf16)v.z, (__bf16)v.w};
  *(uint2*)(dst + off) = *(const uint2*)o4;
}

// ---- Kernel 1: Q/K projections + BN + ReLU, stored transposed (C x DQ) ------
// Qt[b][c][o] = relu(bn1(sum_n Wq[o,n] * x[b,c,n])), Kt likewise with Wk/bn2.
__global__ void qk_proj_kernel(const __bf16* __restrict__ Xb,
                               const __bf16* __restrict__ Wqb, const __bf16* __restrict__ Wkb,
                               const float* __restrict__ g1, const float* __restrict__ b1,
                               const float* __restrict__ m1, const float* __restrict__ v1,
                               const float* __restrict__ g2, const float* __restrict__ b2,
                               const float* __restrict__ m2, const float* __restrict__ v2,
                               __bf16* __restrict__ Qt, __bf16* __restrict__ Kt) {
  const int b = blockIdx.y;
  const int cchunk = blockIdx.x;           // 4 chunks of 256 c-columns
  const int wave = threadIdx.x >> 5;
  const int l = threadIdx.x & 31;
  const int hi = l >> 4;
  const __bf16* xb = Xb + (size_t)b * CC * NN;

  for (int t = wave; t < 128; t += 8) {    // 2 proj * 4 o-tiles * 16 c-subtiles
    const int p  = t >> 6;                 // 0=Q, 1=K
    const int ot = (t >> 4) & 3;
    const int c0 = cchunk * 256 + (t & 15) * 16;
    const __bf16* W = p ? Wkb : Wqb;
    v8f acc = {};
#pragma unroll
    for (int ks = 0; ks < 8; ++ks) {       // K-dim = N = 256
      v16bf a  = fragA_bf16(W + (size_t)ot * 16 * NN + ks * 32, NN);
      v16bf bf = fragB_bf16(xb + (size_t)c0 * NN + ks * 32, NN);  // Bt[c][n] = x row
      acc = wmma_bf16(a, bf, acc);
    }
    const int c = c0 + (l & 15);
    const int obase = ot * 16 + 8 * hi;
    const float* gg = p ? g2 : g1;  const float* be = p ? b2 : b1;
    const float* mm = p ? m2 : m1;  const float* vv = p ? v2 : v1;
    __bf16 o8[8];
#pragma unroll
    for (int r = 0; r < 8; ++r) {
      const int o = obase + r;
      const float sc = gg[o] * rsqrtf(vv[o] + EPSF);
      o8[r] = (__bf16)fmaxf(acc[r] * sc + (be[o] - mm[o] * sc), 0.0f);
    }
    __bf16* dst = (p ? Kt : Qt) + ((size_t)b * CC + c) * DQH + obase;
    *(uint4*)dst = *(const uint4*)o8;      // 8 contiguous bf16 = 16B store
  }
}

// ---- Kernel 2: V projection, stored transposed Vt[b][n][d] ------------------
// Vt[n][d] = relu(bn3(sum_c Wv[d,c] * x[b,c,n]))
__global__ void v_proj_kernel(const __bf16* __restrict__ Xb,
                              const __bf16* __restrict__ Wvb,
                              const float* __restrict__ g3, const float* __restrict__ b3,
                              const float* __restrict__ m3, const float* __restrict__ v3,
                              __bf16* __restrict__ Vt) {
  extern __shared__ __bf16 xT[];           // [64][XLD] transposed bf16 x-block
  const int XLD = CC + 16;
  const int b = blockIdx.y;
  const int nblk = blockIdx.x;             // 4 blocks of 64 n-columns
  const int tid = threadIdx.x, wave = tid >> 5, l = tid & 31, hi = l >> 4;

  // Stage xT[n][c] = x[b][c][nblk*64+n] (transpose via LDS)
  const __bf16* xb = Xb + (size_t)b * CC * NN + nblk * 64;
  for (int c = tid; c < CC; c += 256) {
    const __bf16* row = xb + (size_t)c * NN;
#pragma unroll
    for (int n = 0; n < 64; ++n) xT[(size_t)n * XLD + c] = row[n];
  }
  __syncthreads();

  for (int i = 0; i < 8; ++i) {
    const int mt = wave + 8 * i;           // d-tile 0..63
    v8f acc[4] = {{}, {}, {}, {}};
    for (int ks = 0; ks < 32; ++ks) {      // K-dim = C = 1024
      v16bf a = fragA_bf16(Wvb + (size_t)mt * 16 * CC + ks * 32, CC);
#pragma unroll
      for (int nt = 0; nt < 4; ++nt) {
        v16bf bf = fragB_bf16(xT + (size_t)nt * 16 * XLD + ks * 32, XLD);
        acc[nt] = wmma_bf16(a, bf, acc[nt]);
      }
    }
    const int dbase = mt * 16 + 8 * hi;
    float sc[8], sh[8];
#pragma unroll
    for (int r = 0; r < 8; ++r) {
      const int d = dbase + r;
      sc[r] = g3[d] * rsqrtf(v3[d] + EPSF);
      sh[r] = b3[d] - m3[d] * sc[r];
    }
#pragma unroll
    for (int nt = 0; nt < 4; ++nt) {
      const int n = nblk * 64 + nt * 16 + (l & 15);
      __bf16 o8[8];
#pragma unroll
      for (int r = 0; r < 8; ++r) o8[r] = (__bf16)fmaxf(acc[nt][r] * sc[r] + sh[r], 0.0f);
      *(uint4*)(Vt + ((size_t)b * NN + n) * CC + dbase) = *(const uint4*)o8;
    }
  }
}

// ---- Kernel 3: fused sim -> softmax -> aff @ V -> residual ------------------
// Per workgroup: one batch, 32 rows of the 1024x1024 affinity. sim held in LDS.
__global__ void attn_kernel(const float* __restrict__ x,
                            const __bf16* __restrict__ Qt, const __bf16* __restrict__ Kt,
                            const __bf16* __restrict__ Vt,
                            const float* __restrict__ alpha,
                            float* __restrict__ out) {
  extern __shared__ char smem[];
  float*  simL = (float*)smem;                         // [32][1024] f32
  __bf16* affL = (__bf16*)(smem + 32 * CC * 4);        // [32][ALD] bf16
  const int ALD = CC + 16;
  const int b = blockIdx.y;
  const int cblk = blockIdx.x;                         // 32 rows per block
  const int tid = threadIdx.x, wave = tid >> 5, l = tid & 31, hi = l >> 4;
  const int mt = wave & 1;

  // Phase 1: sim[c,d] = sum_k Kt[c,k] * Qt[d,k]  (K-dim = DQ = 64)
  {
    const __bf16* Abase = Kt + ((size_t)b * CC + cblk * 32 + mt * 16) * DQH;
    const v16bf a0 = fragA_bf16(Abase + 0,  DQH);
    const v16bf a1 = fragA_bf16(Abase + 32, DQH);
    for (int nt = (wave >> 1); nt < 64; nt += 4) {
      const __bf16* Bbase = Qt + ((size_t)b * CC + (size_t)nt * 16) * DQH;
      v8f acc = {};
      acc = wmma_bf16(a0, fragB_bf16(Bbase + 0,  DQH), acc);
      acc = wmma_bf16(a1, fragB_bf16(Bbase + 32, DQH), acc);
      const int d = nt * 16 + (l & 15);
      const int mrow = mt * 16 + 8 * hi;
#pragma unroll
      for (int r = 0; r < 8; ++r) simL[(size_t)(mrow + r) * CC + d] = acc[r];
    }
  }
  __syncthreads();

  // Phase 2: softmax(max - sim) == exp(rowmin - sim)/sum. One wave per 4 rows.
  for (int rr = 0; rr < 4; ++rr) {
    const int row = wave * 4 + rr;
    const float* srow = simL + (size_t)row * CC;
    float mn = 3.4e38f;
    for (int d = l; d < CC; d += 32) mn = fminf(mn, srow[d]);
#pragma unroll
    for (int off = 16; off > 0; off >>= 1) mn = fminf(mn, __shfl_xor(mn, off, 32));
    float s = 0.0f;
    for (int d = l; d < CC; d += 32) s += __expf(mn - srow[d]);
#pragma unroll
    for (int off = 16; off > 0; off >>= 1) s += __shfl_xor(s, off, 32);
    const float inv = 1.0f / s;
    __bf16* arow = affL + (size_t)row * ALD;
    for (int d = l; d < CC; d += 32) arow[d] = (__bf16)(__expf(mn - srow[d]) * inv);
  }
  __syncthreads();

  // Phase 3: out = alpha * (aff @ V) + x.  A from LDS aff, B from Vt (k-contig).
  const float al = alpha[0];
  for (int nt = (wave >> 1); nt < 16; nt += 4) {
    const __bf16* Bbase = Vt + ((size_t)b * NN + (size_t)nt * 16) * CC;
    v8f acc = {};
    for (int ks = 0; ks < 32; ++ks) {      // K-dim = C = 1024
      __builtin_prefetch(Bbase + (ks + 1) * 32, 0, 1);
      v16bf a  = fragA_bf16(affL + (size_t)mt * 16 * ALD + ks * 32, ALD);
      v16bf bf = fragB_bf16(Bbase + ks * 32, CC);
      acc = wmma_bf16(a, bf, acc);
    }
    const int n = nt * 16 + (l & 15);
    const int c0 = cblk * 32 + mt * 16 + 8 * hi;
#pragma unroll
    for (int r = 0; r < 8; ++r) {
      const size_t idx = ((size_t)b * CC + (size_t)(c0 + r)) * NN + n;
      out[idx] = al * acc[r] + x[idx];
    }
  }
}

extern "C" void kernel_launch(void* const* d_in, const int* in_sizes, int n_in,
                              void* d_out, int out_size, void* d_ws, size_t ws_size,
                              hipStream_t stream) {
  const float* x  = (const float*)d_in[0];
  const float* Wq = (const float*)d_in[1];
  const float* Wk = (const float*)d_in[2];
  const float* Wv = (const float*)d_in[3];
  const float* g1 = (const float*)d_in[4],  *b1 = (const float*)d_in[5];
  const float* m1 = (const float*)d_in[6],  *v1 = (const float*)d_in[7];
  const float* g2 = (const float*)d_in[8],  *b2 = (const float*)d_in[9];
  const float* m2 = (const float*)d_in[10], *v2 = (const float*)d_in[11];
  const float* g3 = (const float*)d_in[12], *b3 = (const float*)d_in[13];
  const float* m3 = (const float*)d_in[14], *v3 = (const float*)d_in[15];
  const float* alpha = (const float*)d_in[16];
  float* out = (float*)d_out;

  // Workspace (bf16): Qt 4MB | Kt 4MB | Vt 16MB | Xb 16MB | Wvb 2MB | Wqb | Wkb
  __bf16* Qt  = (__bf16*)d_ws;
  __bf16* Kt  = Qt  + (size_t)BB * CC * DQH;
  __bf16* Vt  = Kt  + (size_t)BB * CC * DQH;
  __bf16* Xb  = Vt  + (size_t)BB * NN * CC;
  __bf16* Wvb = Xb  + XE;
  __bf16* Wqb = Wvb + WVE;
  __bf16* Wkb = Wqb + WQE;

  const size_t cvt4 = (XE + WVE + 2 * WQE) / 4;          // 4 elems per thread
  cvt_kernel<<<dim3((unsigned)((cvt4 + 255) / 256)), 256, 0, stream>>>(
      x, Wv, Wq, Wk, Xb, Wvb, Wqb, Wkb);

  qk_proj_kernel<<<dim3(4, BB), 256, 0, stream>>>(
      Xb, Wqb, Wkb, g1, b1, m1, v1, g2, b2, m2, v2, Qt, Kt);

  const size_t lds2 = (size_t)64 * (CC + 16) * sizeof(__bf16);       // ~130 KB
  v_proj_kernel<<<dim3(4, BB), 256, lds2, stream>>>(
      Xb, Wvb, g3, b3, m3, v3, Vt);

  const size_t lds3 = (size_t)32 * CC * 4 + (size_t)32 * (CC + 16) * 2;  // ~193 KB
  attn_kernel<<<dim3(CC / 32, BB), 256, lds3, stream>>>(
      x, Qt, Kt, Vt, alpha, out);
}
